// ASAPClassifer_79843442033169
// MI455X (gfx1250) — matrix operand order repaired and checked by reading, not compiled, not run on hardware
//
#include <hip/hip_runtime.h>
#include <cstdint>
#include <cstddef>

#define N0_   20000
#define INCH_ 64
#define HID_  512
#define E0_   120000

typedef __attribute__((ext_vector_type(16))) __bf16 v16bf;
typedef __attribute__((ext_vector_type(8)))  __bf16 v8bf;
typedef __attribute__((ext_vector_type(8)))  float  v8f;
typedef int gv4i __attribute__((vector_size(16)));   // matches builtin param pointee

#define GAS __attribute__((address_space(1)))
#define LAS __attribute__((address_space(3)))

// ---------------- CDNA5 async global->LDS helpers ----------------

__device__ __forceinline__ void async_copy_b128(const void* g, void* l) {
#if __has_builtin(__builtin_amdgcn_global_load_async_to_lds_b128)
    __builtin_amdgcn_global_load_async_to_lds_b128((GAS gv4i*)g, (LAS gv4i*)l, 0, 0);
#else
    *(v8bf*)l = *(const v8bf*)g;   // fallback: synchronous copy through VGPRs
#endif
}

template <int CNT>
__device__ __forceinline__ void wait_async_le() {
#if __has_builtin(__builtin_amdgcn_global_load_async_to_lds_b128)
 #if __has_builtin(__builtin_amdgcn_s_wait_asynccnt)
    __builtin_amdgcn_s_wait_asynccnt(CNT);
 #else
    asm volatile("s_wait_asynccnt %0" :: "i"(CNT) : "memory");
 #endif
#endif
}

// ---------------- utility device helpers ----------------

__device__ __forceinline__ unsigned fkey(float f) {
    unsigned u = __float_as_uint(f);
    return (u & 0x80000000u) ? ~u : (u | 0x80000000u);
}
__device__ __forceinline__ float finvkey(unsigned k) {
    unsigned u = (k & 0x80000000u) ? (k ^ 0x80000000u) : ~k;
    return __uint_as_float(u);
}
// edges augmented with self loops: e in [0,E) real, e in [E,E+n) self
__device__ __forceinline__ void edge_sd(int e, const int* src, const int* dst, int E,
                                        int& s, int& d) {
    if (e < E) { s = src[e]; d = dst[e]; } else { s = e - E; d = e - E; }
}

// ---------------- small kernels ----------------

__global__ void fill_f32(float* p, float v, long total) {
    long i = (long)blockIdx.x * blockDim.x + threadIdx.x;
    if (i < total) p[i] = v;
}

__global__ void cvt_f32_bf16(const float* __restrict__ in, __bf16* __restrict__ out, long total) {
    long i = (long)blockIdx.x * blockDim.x + threadIdx.x;
    if (i < total) out[i] = (__bf16)in[i];
}

// out[n][k] = (bf16) in[k][n]   (K x N fp32 -> N x K bf16)
__global__ void cvt_transpose_bf16(const float* __restrict__ in, __bf16* __restrict__ out,
                                   int K, int N) {
    long idx = (long)blockIdx.x * blockDim.x + threadIdx.x;
    if (idx >= (long)K * N) return;
    int k = (int)(idx / N), nn = (int)(idx % N);
    out[(long)nn * K + k] = (__bf16)in[idx];
}

__global__ void keys_to_float(unsigned* p, long total) {
    long i = (long)blockIdx.x * blockDim.x + threadIdx.x;
    if (i < total) ((float*)p)[i] = finvkey(p[i]);
}

// ---------------- graph scatter kernels ----------------

__global__ void edge_agg(const float* __restrict__ x, const int* __restrict__ src,
                         const int* __restrict__ dst, int E, int F, float* agg) {
    long idx = (long)blockIdx.x * blockDim.x + threadIdx.x;
    if (idx >= (long)E * F) return;
    int e = (int)(idx / F), f = (int)(idx % F);
    atomicAdd(&agg[(long)dst[e] * F + f], x[(long)src[e] * F + f]);
}

__global__ void deg_count(const int* __restrict__ dst, int E, float* deg) {
    int e = blockIdx.x * blockDim.x + threadIdx.x;
    if (e < E) atomicAdd(&deg[dst[e]], 1.0f);
}

__global__ void finalize_mean(float* agg, const float* __restrict__ deg, int n, int F) {
    long idx = (long)blockIdx.x * blockDim.x + threadIdx.x;
    if (idx >= (long)n * F) return;
    int i = (int)(idx / F);
    agg[idx] = agg[idx] / fmaxf(deg[i], 1.0f);
}

__global__ void edge_segmax(const float* __restrict__ h, const int* __restrict__ src,
                            const int* __restrict__ dst, int E, int n, int F, unsigned* keys) {
    long idx = (long)blockIdx.x * blockDim.x + threadIdx.x;
    if (idx >= (long)(E + n) * F) return;
    int e = (int)(idx / F), f = (int)(idx % F);
    int s, d; edge_sd(e, src, dst, E, s, d);
    atomicMax(&keys[(long)d * F + f], fkey(h[(long)s * F + f]));
}

__global__ void edge_score(const float* __restrict__ q, const float* __restrict__ r,
                           const float* __restrict__ ab, int li,
                           const int* __restrict__ src, const int* __restrict__ dst,
                           int E, int n, float* sc, unsigned* smaxk) {
    int e = blockIdx.x * blockDim.x + threadIdx.x;
    if (e >= E + n) return;
    int s, d; edge_sd(e, src, dst, E, s, d);
    float v = q[d] + r[s] + ab[li];
    v = (v > 0.0f) ? v : 0.2f * v;          // leaky_relu(0.2)
    sc[e] = v;
    atomicMax(&smaxk[d], fkey(v));
}

__global__ void edge_exp(const float* __restrict__ sc, const unsigned* __restrict__ smaxk,
                         const int* __restrict__ src, const int* __restrict__ dst,
                         int E, int n, float* ex, float* denom) {
    int e = blockIdx.x * blockDim.x + threadIdx.x;
    if (e >= E + n) return;
    int s, d; edge_sd(e, src, dst, E, s, d);
    float v = __expf(sc[e] - finvkey(smaxk[d]));
    ex[e] = v;
    atomicAdd(&denom[d], v);
}

__global__ void edge_xn(const float* __restrict__ h, const float* __restrict__ ex,
                        const float* __restrict__ denom, const int* __restrict__ src,
                        const int* __restrict__ dst, int E, int n, int F, float* xn) {
    long idx = (long)blockIdx.x * blockDim.x + threadIdx.x;
    if (idx >= (long)(E + n) * F) return;
    int e = (int)(idx / F), f = (int)(idx % F);
    int s, d; edge_sd(e, src, dst, E, s, d);
    float att = ex[e] / denom[d];
    atomicAdd(&xn[(long)d * F + f], h[(long)s * F + f] * att);
}

__global__ void edge_leconv(const float* __restrict__ a, const float* __restrict__ b2,
                            const int* __restrict__ src, const int* __restrict__ dst,
                            int E, int n, float* aggf) {
    int e = blockIdx.x * blockDim.x + threadIdx.x;
    if (e >= E + n) return;
    int s, d; edge_sd(e, src, dst, E, s, d);
    atomicAdd(&aggf[d], a[s] - b2[d]);
}

__global__ void fit_kernel(const float* __restrict__ aggf, const float* __restrict__ c,
                           float* fit, int n) {
    int i = blockIdx.x * blockDim.x + threadIdx.x;
    if (i < n) fit[i] = 1.0f / (1.0f + __expf(-(aggf[i] + c[i])));
}

// ---------------- per-node row dot (wave per node, shuffle reduce) ----------------

__global__ __launch_bounds__(256)
void rowdot_kernel(const float* __restrict__ X, const float* __restrict__ w,
                   const float* __restrict__ bias, int bidx, float* out, int n, int F) {
    int wv = threadIdx.x >> 5, lane = threadIdx.x & 31;
    int node = blockIdx.x * 8 + wv;
    if (node >= n) return;
    float s = 0.0f;
    for (int j = lane; j < F; j += 32) s += X[(long)node * F + j] * w[j];
    #pragma unroll
    for (int o = 16; o > 0; o >>= 1) s += __shfl_xor(s, o, 32);
    if (lane == 0) out[node] = s + (bias ? bias[bidx] : 0.0f);
}

// ---------------- top-k by exact rank counting ----------------

__global__ __launch_bounds__(256)
void rank_kernel(const float* __restrict__ fit, int* rank, int n, int kkeep) {
    int i = blockIdx.x;
    float fi = fit[i];
    __shared__ int cnt_s;
    if (threadIdx.x == 0) cnt_s = 0;
    __syncthreads();
    int c = 0;
    for (int j = threadIdx.x; j < n; j += 256) {
        float fj = fit[j];
        if (fj > fi || (fj == fi && j < i)) ++c;
    }
    atomicAdd(&cnt_s, c);
    __syncthreads();
    if (threadIdx.x == 0) rank[i] = (cnt_s < kkeep) ? cnt_s : -1;
}

__global__ void gather_kernel(const float* __restrict__ xn, const float* __restrict__ fit,
                              const float* __restrict__ pos, const int* __restrict__ rank,
                              float* xout, float* posout, int n, int F) {
    long idx = (long)blockIdx.x * blockDim.x + threadIdx.x;
    if (idx >= (long)n * F) return;
    int i = (int)(idx / F), f = (int)(idx % F);
    int r = rank[i];
    if (r < 0) return;
    xout[(long)r * F + f] = xn[idx] * fit[i];
    if (f < 2) posout[r * 2 + f] = pos[i * 2 + f];
}

__global__ void colmax_kernel(const float* __restrict__ X, float* out, int n, int F) {
    int c = blockIdx.x * blockDim.x + threadIdx.x;
    if (c >= F) return;
    float m = -3.4e38f;
    for (int i = 0; i < n; ++i) m = fmaxf(m, X[(long)i * F + c]);
    out[c] = m;
}

// ---------------- brute-force kNN (per-thread insertion top-k) ----------------

__global__ void knn_kernel(const float* __restrict__ pos, int n, int k, int* ksrc, int* kdst) {
    int i = blockIdx.x * blockDim.x + threadIdx.x;
    if (i >= n) return;
    float xi = pos[2 * i], yi = pos[2 * i + 1];
    float bd[8]; int bi[8];
    #pragma unroll
    for (int t = 0; t < 8; ++t) { bd[t] = 3.4e38f; bi[t] = 0; }
    for (int j = 0; j < n; ++j) {
        if (j == i) continue;
        float dx = pos[2 * j] - xi, dy = pos[2 * j + 1] - yi;
        float d = dx * dx + dy * dy;
        if (d < bd[k - 1]) {
            int t = k - 1;
            while (t > 0 && bd[t - 1] > d) { bd[t] = bd[t - 1]; bi[t] = bi[t - 1]; --t; }
            bd[t] = d; bi[t] = j;
        }
    }
    for (int t = 0; t < k; ++t) { ksrc[(long)i * k + t] = bi[t]; kdst[(long)i * k + t] = i; }
}

// ---------------- WMMA bf16 GEMM: C = A@Wt^T [+bias][+Cin][relu] ----------------
// A: M x K bf16 row-major.  Wt: N x K bf16 (pre-transposed weights).
// Block: 256 threads = 8 wave32. Tile BM=64, BN=128, BK=32.
// Triple-buffered async pipeline: tile t+1 is staged with
// GLOBAL_LOAD_ASYNC_TO_LDS_B128 while tile t is computed; s_wait_asynccnt 3
// drains only the older tile's 3 transfers. One barrier per K-step.
// Each wave computes 16 rows x 64 cols = 4 WMMAs sharing one A fragment.
#define BM 64
#define BN 128
#define BK 32
#define ASTR 40   // bf16 elements per LDS row (80B: 16B-aligned, conflict-free starts)
#define BSTR 40
#define GF_BIAS 1
#define GF_CIN  2
#define GF_RELU 4

__global__ __launch_bounds__(256)
void gemm_bf16_wmma(const __bf16* __restrict__ A, const __bf16* __restrict__ Wt,
                    const float* __restrict__ bias, const float* __restrict__ Cin,
                    float* __restrict__ C, int M, int N, int K, int flags) {
    __shared__ __bf16 sA[3][BM * ASTR];   // A tiles, stored [row][k], unswizzled
    __shared__ __bf16 sB[3][BN * BSTR];   // B tiles, stored [col][k]

    const int tid  = threadIdx.x;
    const int wv   = tid >> 5;
    const int lane = tid & 31;
    const int wm   = wv >> 1;        // 0..3 : 16-row slab
    const int wn   = wv & 1;         // 0..1 : 64-col slab (4 x 16-col subtiles)
    const int rowBase = blockIdx.y * BM;
    const int colBase = blockIdx.x * BN;
    const int hh  = lane >> 4;
    const int l16 = lane & 15;

    // staging coordinates: A = one 16B transfer/thread, B = two 16B transfers/thread
    const int ar   = tid >> 2;                               // A row 0..63
    const int ac8  = (tid & 3) * 8;                          // A k-offset {0,8,16,24}
    const int apos = ac8;                                    // unswizzled k-offset
    const int bcol = tid >> 2;                               // B col 0..63 (and +64)
    const int bk8  = (tid & 3) * 8;                          // B k-offset {0,8,16,24}

    auto stage = [&](int k0, int buf) {
        int gr = rowBase + ar;
        __bf16* al = &sA[buf][ar * ASTR + apos];
        if (gr < M) {
            async_copy_b128(&A[(long)gr * K + k0 + ac8], al);
        } else {
            v8bf z = {};
            *(v8bf*)al = z;
        }
        async_copy_b128(&Wt[(long)(colBase + bcol) * K + k0 + bk8],
                        &sB[buf][bcol * BSTR + bk8]);
        async_copy_b128(&Wt[(long)(colBase + bcol + 64) * K + k0 + bk8],
                        &sB[buf][(bcol + 64) * BSTR + bk8]);
    };

    v8f acc[4];
    #pragma unroll
    for (int s = 0; s < 4; ++s) acc[s] = (v8f){};

    const int ntile = K / BK;
    stage(0, 0);

    for (int t = 0; t < ntile; ++t) {
        const int buf = t % 3;
        if (t + 1 < ntile) {
            stage((t + 1) * BK, (t + 1) % 3);
            wait_async_le<3>();     // drain tile t; keep tile t+1 in flight
        } else {
            wait_async_le<0>();
        }
        __syncthreads();            // tile t visible to all waves (single barrier/step)

        // ---- A fragment: 16 contiguous bf16 -> 2 x ds_load_b128 ----
        const __bf16* ap = &sA[buf][(wm * 16 + l16) * ASTR + hh * 16];
        v8bf alo = *(const v8bf*)ap;
        v8bf ahi = *(const v8bf*)(ap + 8);
        v16bf af = __builtin_shufflevector(alo, ahi,
                    0, 1, 2, 3, 4, 5, 6, 7, 8, 9, 10, 11, 12, 13, 14, 15);

        // ---- 4 B subtiles (16 cols each): K = hh*16 + e contiguous ----
        #pragma unroll
        for (int s = 0; s < 4; ++s) {
            const __bf16* bp = &sB[buf][(wn * 64 + s * 16 + l16) * BSTR + hh * 16];
            v8bf blo = *(const v8bf*)bp;
            v8bf bhi = *(const v8bf*)(bp + 8);
            v16bf bfv = __builtin_shufflevector(blo, bhi,
                        0, 1, 2, 3, 4, 5, 6, 7, 8, 9, 10, 11, 12, 13, 14, 15);
            acc[s] = __builtin_amdgcn_wmma_f32_16x16x32_bf16(false, af, false, bfv,
                                                             (short)0, acc[s], false, false);
        }
        // No trailing barrier: triple buffering guarantees the buffer read at step t
        // is only overwritten at step t+2, after everyone passed barrier t+1.
    }

    // epilogue: C/D 16x16 f32 layout — VGPR r holds M = r + 8*hh
    const int mofs = hh * 8;
    #pragma unroll
    for (int s = 0; s < 4; ++s) {
        int c = colBase + wn * 64 + s * 16 + l16;
        #pragma unroll
        for (int r = 0; r < 8; ++r) {
            int gr = rowBase + wm * 16 + mofs + r;
            if (gr < M) {
                float v = acc[s][r];
                if (flags & GF_BIAS) v += bias[c];
                if (flags & GF_CIN)  v += Cin[(long)gr * N + c];
                if (flags & GF_RELU) v = fmaxf(v, 0.0f);
                C[(long)gr * N + c] = v;
            }
        }
    }
}

// ---------------- MLP head (single block) ----------------

__global__ __launch_bounds__(512)
void mlp_head(const float* __restrict__ xs, const float* __restrict__ w1,
              const float* __restrict__ b1, const float* __restrict__ w2,
              const float* __restrict__ b2, float* out) {
    __shared__ float hid[HID_];
    int t = threadIdx.x;
    float s = b1[t];
    for (int j = 0; j < 3 * HID_; ++j) s += xs[j] * w1[(long)j * HID_ + t];
    hid[t] = fmaxf(s, 0.0f);
    __syncthreads();
    if (t < 10) {
        float o = b2[t];
        for (int j = 0; j < HID_; ++j) o += hid[j] * w2[j * 10 + t];
        out[t] = o;
    }
}

// ---------------- host orchestration ----------------

static inline unsigned nblk(long total, int bs) { return (unsigned)((total + bs - 1) / bs); }

extern "C" void kernel_launch(void* const* d_in, const int* in_sizes, int n_in,
                              void* d_out, int out_size, void* d_ws, size_t ws_size,
                              hipStream_t stream) {
    const float* x_in   = (const float*)d_in[0];
    const float* pos_in = (const float*)d_in[1];
    const int*   ei     = (const int*)d_in[2];
    const float* c0wr = (const float*)d_in[3];
    const float* c0br = (const float*)d_in[4];
    const float* c0wl = (const float*)d_in[5];
    const float* cwr  = (const float*)d_in[6];
    const float* cbr  = (const float*)d_in[7];
    const float* cwl  = (const float*)d_in[8];
    const float* plw  = (const float*)d_in[9];
    const float* plb  = (const float*)d_in[10];
    const float* paw  = (const float*)d_in[11];
    const float* pab  = (const float*)d_in[12];
    const float* l1w  = (const float*)d_in[13];
    const float* l1b  = (const float*)d_in[14];
    const float* l2w  = (const float*)d_in[15];
    const float* l3w  = (const float*)d_in[16];
    const float* l3b  = (const float*)d_in[17];
    const float* w1   = (const float*)d_in[18];
    const float* b1   = (const float*)d_in[19];
    const float* w2   = (const float*)d_in[20];
    const float* b2   = (const float*)d_in[21];

    // carve workspace
    char* base = (char*)d_ws;
    size_t off = 0;
    auto carve = [&](size_t bytes) -> void* {
        void* p = base + off;
        off += (bytes + 255) & ~(size_t)255;
        return p;
    };
    float*    F_h    = (float*)carve((size_t)N0_ * HID_ * 4);
    float*    F_mean = (float*)carve((size_t)N0_ * HID_ * 4);  // also xq_out
    float*    F_xn   = (float*)carve((size_t)N0_ * HID_ * 4);  // also segmax keys
    float*    F_xc   = (float*)carve((size_t)N0_ * HID_ * 4);  // pooled x for next layer
    __bf16*   Abf    = (__bf16*)carve((size_t)N0_ * HID_ * 2);
    __bf16*   WbfT   = (__bf16*)carve((size_t)HID_ * HID_ * 2);
    float*    deg    = (float*)carve((size_t)N0_ * 4);
    float*    qb     = (float*)carve((size_t)N0_ * 4);
    float*    rb     = (float*)carve((size_t)N0_ * 4);
    float*    le_a   = (float*)carve((size_t)N0_ * 4);
    float*    le_b   = (float*)carve((size_t)N0_ * 4);
    float*    le_c   = (float*)carve((size_t)N0_ * 4);
    float*    aggf   = (float*)carve((size_t)N0_ * 4);
    float*    fitv   = (float*)carve((size_t)N0_ * 4);
    float*    denom  = (float*)carve((size_t)N0_ * 4);
    unsigned* smaxk  = (unsigned*)carve((size_t)N0_ * 4);
    float*    sc     = (float*)carve((size_t)(E0_ + N0_) * 4);
    float*    exv    = (float*)carve((size_t)(E0_ + N0_) * 4);
    int*      rankb  = (int*)carve((size_t)N0_ * 4);
    float*    posA   = (float*)carve((size_t)N0_ * 2 * 4);
    float*    posB   = (float*)carve((size_t)N0_ * 2 * 4);
    int*      ksrc   = (int*)carve((size_t)80000 * 4);
    int*      kdst   = (int*)carve((size_t)80000 * 4);
    float*    xs     = (float*)carve((size_t)3 * HID_ * 4);

    auto FILL = [&](float* p, long total) {
        fill_f32<<<nblk(total, 256), 256, 0, stream>>>(p, 0.0f, total);
    };
    auto CVT = [&](const float* src, __bf16* dst, long total) {
        cvt_f32_bf16<<<nblk(total, 256), 256, 0, stream>>>(src, dst, total);
    };
    auto CVTT = [&](const float* w, int K, int N) {   // -> WbfT[N][K]
        cvt_transpose_bf16<<<nblk((long)K * N, 256), 256, 0, stream>>>(w, WbfT, K, N);
    };
    auto GEMM = [&](const __bf16* A, const float* bias, const float* Cin, float* Cp,
                    int M, int N, int K, int flags) {
        dim3 grid(N / BN, (M + BM - 1) / BM);
        gemm_bf16_wmma<<<grid, 256, 0, stream>>>(A, WbfT, bias, Cin, Cp, M, N, K, flags);
    };
    auto ROWDOT = [&](const float* X, const float* w, const float* bias, int bidx,
                      float* out, int n) {
        rowdot_kernel<<<nblk((long)n, 8), 256, 0, stream>>>(X, w, bias, bidx, out, n, HID_);
    };

    int n = N0_;
    const float* xcur = x_in;
    const float* pcur = pos_in;
    const int* esrc = ei;
    const int* edst = ei + E0_;
    int E = E0_;

    for (int li = 0; li < 3; ++li) {
        const int K = (li == 0) ? INCH_ : HID_;
        const float* wr = (li == 0) ? c0wr : cwr + (size_t)(li - 1) * HID_ * HID_;
        const float* br = (li == 0) ? c0br : cbr + (size_t)(li - 1) * HID_;
        const float* wl = (li == 0) ? c0wl : cwl + (size_t)(li - 1) * HID_ * HID_;

        // --- GraphConv: h = relu(mean@wr + br + x@wl) ---
        FILL(F_mean, (long)n * K);
        FILL(deg, n);
        edge_agg<<<nblk((long)E * K, 256), 256, 0, stream>>>(xcur, esrc, edst, E, K, F_mean);
        deg_count<<<nblk(E, 256), 256, 0, stream>>>(edst, E, deg);
        finalize_mean<<<nblk((long)n * K, 256), 256, 0, stream>>>(F_mean, deg, n, K);
        CVT(F_mean, Abf, (long)n * K);
        CVTT(wr, K, HID_);
        GEMM(Abf, br, nullptr, F_h, n, HID_, K, GF_BIAS);
        CVT(xcur, Abf, (long)n * K);
        CVTT(wl, K, HID_);
        GEMM(Abf, nullptr, F_h, F_h, n, HID_, K, GF_CIN | GF_RELU);

        // --- ASAP pool ---
        // xq = lin(segment_max(h over edges+self))
        FILL(F_xn, (long)n * HID_);  // zero keys
        edge_segmax<<<nblk((long)(E + n) * HID_, 256), 256, 0, stream>>>(
            F_h, esrc, edst, E, n, HID_, (unsigned*)F_xn);
        keys_to_float<<<nblk((long)n * HID_, 256), 256, 0, stream>>>((unsigned*)F_xn, (long)n * HID_);
        CVT(F_xn, Abf, (long)n * HID_);
        CVTT(plw + (size_t)li * HID_ * HID_, HID_, HID_);
        GEMM(Abf, plb + (size_t)li * HID_, nullptr, F_mean, n, HID_, HID_, GF_BIAS);

        // attention score projections, segment softmax, xn
        ROWDOT(F_mean, paw + (size_t)li * 2 * HID_, nullptr, 0, qb, n);
        ROWDOT(F_h,    paw + (size_t)li * 2 * HID_ + HID_, nullptr, 0, rb, n);
        FILL((float*)smaxk, n);
        FILL(denom, n);
        edge_score<<<nblk(E + n, 256), 256, 0, stream>>>(qb, rb, pab, li, esrc, edst, E, n, sc, smaxk);
        edge_exp<<<nblk(E + n, 256), 256, 0, stream>>>(sc, smaxk, esrc, edst, E, n, exv, denom);
        FILL(F_xn, (long)n * HID_);
        edge_xn<<<nblk((long)(E + n) * HID_, 256), 256, 0, stream>>>(
            F_h, exv, denom, esrc, edst, E, n, HID_, F_xn);

        // LEConv fitness
        ROWDOT(F_xn, l1w + (size_t)li * HID_, l1b, li, le_a, n);
        ROWDOT(F_xn, l2w + (size_t)li * HID_, nullptr, 0, le_b, n);
        ROWDOT(F_xn, l3w + (size_t)li * HID_, l3b, li, le_c, n);
        FILL(aggf, n);
        edge_leconv<<<nblk(E + n, 256), 256, 0, stream>>>(le_a, le_b, esrc, edst, E, n, aggf);
        fit_kernel<<<nblk(n, 256), 256, 0, stream>>>(aggf, le_c, fitv, n);

        // top-k selection + gather
        int kk = (n + 1) / 2;
        rank_kernel<<<n, 256, 0, stream>>>(fitv, rankb, n, kk);
        float* pnext = (li & 1) ? posB : posA;
        gather_kernel<<<nblk((long)n * HID_, 256), 256, 0, stream>>>(
            F_xn, fitv, pcur, rankb, F_xc, pnext, n, HID_);

        // global max pool readout for this layer
        colmax_kernel<<<nblk(HID_, 256), 256, 0, stream>>>(F_xc, xs + (size_t)li * HID_, kk, HID_);

        // rebuild kNN graph for next layer
        if (li < 2) {
            int kNN = 6 + 2 * li;
            knn_kernel<<<nblk(kk, 256), 256, 0, stream>>>(pnext, kk, kNN, ksrc, kdst);
            esrc = ksrc;
            edst = kdst;
            E = kk * kNN;
        }
        xcur = F_xc;
        pcur = pnext;
        n = kk;
    }

    // MLP head -> d_out (10 floats)
    mlp_head<<<1, HID_, 0, stream>>>(xs, w1, b1, w2, b2, (float*)d_out);
}
